// GraphLSTMModel_1477468750567
// MI455X (gfx1250) — matrix operand order repaired and checked by reading, not compiled, not run on hardware
//
#include <hip/hip_runtime.h>
#include <hip/hip_bf16.h>
#include <math.h>

typedef __attribute__((ext_vector_type(16))) __bf16 v16bf;
typedef __attribute__((ext_vector_type(8)))  __bf16 v8bf;
typedef __attribute__((ext_vector_type(8)))  float  v8f;

__device__ __forceinline__ float sigm(float x) { return 1.0f / (1.0f + __expf(-x)); }

// ---------------------------------------------------------------- utilities
__global__ __launch_bounds__(256) void zero_i32(int* p, int n) {
    int i = blockIdx.x * 256 + threadIdx.x;
    if (i < n) p[i] = 0;
}
__global__ __launch_bounds__(256) void zero_f32(float* p, int n) {
    int i = blockIdx.x * 256 + threadIdx.x;
    if (i < n) p[i] = 0.0f;
}

// Count in-degree per node
__global__ __launch_bounds__(256) void deg_count(const int* __restrict__ dst, int* __restrict__ deg, int E) {
    int e = blockIdx.x * 256 + threadIdx.x;
    if (e < E) atomicAdd(&deg[dst[e]], 1);
}

// Single-block exclusive scan of degrees -> row_start, plus invdeg = 1/max(deg,1)
__global__ __launch_bounds__(1024) void scan_deg(const int* __restrict__ deg, int* __restrict__ row_start,
                                                 float* __restrict__ invdeg, int n) {
    __shared__ int sums[1024];
    const int tid = threadIdx.x;
    const int chunk = (n + 1023) / 1024;
    const int st = tid * chunk;
    int s = 0;
    for (int i = 0; i < chunk; ++i) {
        int idx = st + i;
        if (idx < n) s += deg[idx];
    }
    sums[tid] = s;
    __syncthreads();
    for (int off = 1; off < 1024; off <<= 1) {
        int v = (tid >= off) ? sums[tid - off] : 0;
        __syncthreads();
        sums[tid] += v;
        __syncthreads();
    }
    int run = sums[tid] - s;  // exclusive prefix
    for (int i = 0; i < chunk; ++i) {
        int idx = st + i;
        if (idx < n) {
            row_start[idx] = run;
            int d = deg[idx];
            run += d;
            invdeg[idx] = 1.0f / fmaxf((float)d, 1.0f);
        }
    }
    if (tid == 1023) row_start[n] = run;
}

// Scatter edges into CSR buckets
__global__ __launch_bounds__(256) void csr_fill(const int* __restrict__ src, const int* __restrict__ dst,
                                                const float* __restrict__ ew, const int* __restrict__ row_start,
                                                int* __restrict__ cursor, int* __restrict__ csrc,
                                                float* __restrict__ cw, int E) {
    int e = blockIdx.x * 256 + threadIdx.x;
    if (e < E) {
        int d = dst[e];
        int pos = atomicAdd(&cursor[d], 1);
        int idx = row_start[d] + pos;
        csrc[idx] = src[e];
        cw[idx]   = ew[e];
    }
}

// ---------------------------------------------------------------- weight packing
// Stack [W1;W2] (f32, [K1,NC];[K2,NC]) into bf16 *fragment-major* layout for WMMA B operands:
// dst[((kbi*CT + ct)*32 + lane)*16 + e] where kbi = K-block of 32, ct = 16-col tile,
// lane's column = ct*16 + (lane&15), element e covers K = kbi*32 + (lane>>4)*16 + e.
// Each lane's 16 bf16 values are then one contiguous, 32B-aligned v16bf load in the GEMM.
__global__ __launch_bounds__(256) void pack_wmma_b(const float* __restrict__ W1, const float* __restrict__ W2,
                                                   __bf16* __restrict__ dstp, int K1, int K2, int NC) {
    int idx = blockIdx.x * 256 + threadIdx.x;
    int K = K1 + K2;
    if (idx >= K * NC) return;
    int e    = idx & 15;
    int lane = (idx >> 4) & 31;
    int frag = idx >> 9;  // kbi*CT + ct
    int CT   = NC / 16;
    int ct   = frag % CT;
    int kbi  = frag / CT;
    int k    = kbi * 32 + (lane >> 4) * 16 + e;
    int col  = ct * 16 + (lane & 15);
    float v  = (k < K1) ? W1[k * NC + col] : W2[(k - K1) * NC + col];
    dstp[idx] = (__bf16)v;
}

// ---------------------------------------------------------------- gathers
// One wave per (t,n); lane = feature (F=32). Edge metadata batch-loaded once per wave
// (one edge per lane) and broadcast via shuffles: feature reads are the only per-edge traffic.
__global__ __launch_bounds__(256) void sage_gather(const int* __restrict__ rs, const int* __restrict__ cs,
                                                   const float* __restrict__ cw, const float* __restrict__ invd,
                                                   const float* __restrict__ x, float* __restrict__ aggs,
                                                   int N, int TN) {
    const int wave = threadIdx.x >> 5, lane = threadIdx.x & 31;
    int p = blockIdx.x * 8 + wave;
    if (p >= TN) return;
    int t = p / N, n = p - t * N;
    const float* xb = x + (size_t)t * N * 32;
    float acc = 0.0f;
    const int b = rs[n], e = rs[n + 1];
    for (int i0 = b; i0 < e; i0 += 32) {
        int ii = i0 + lane;
        int   s_l = (ii < e) ? cs[ii] : 0;
        float w_l = (ii < e) ? cw[ii] : 0.0f;
        int cnt = min(32, e - i0);
        for (int j = 0; j < cnt; ++j) {
            int   s = __shfl(s_l, j, 32);
            float w = __shfl(w_l, j, 32);
            acc += w * xb[(size_t)s * 32 + lane];
        }
    }
    aggs[(size_t)p * 32 + lane] = acc * invd[n];
}

// LSTM aggregation: 4 waves per node cover the 128-wide inp=[xr_t(64), h(64)].
// n is wave-uniform -> same shuffle-broadcast trick.
__global__ __launch_bounds__(256) void lstm_gather(const int* __restrict__ rs, const int* __restrict__ cs,
                                                   const float* __restrict__ cw, const float* __restrict__ invd,
                                                   const float* __restrict__ xr_t, const float* __restrict__ h,
                                                   float* __restrict__ agg, int N) {
    const int wave = threadIdx.x >> 5, lane = threadIdx.x & 31;
    int n = blockIdx.x * 2 + (wave >> 2);
    if (n >= N) return;
    int k = ((wave & 3) << 5) | lane;
    const float* base = (k < 64) ? (xr_t + k) : (h + (k - 64));
    float acc = 0.0f;
    const int b = rs[n], e = rs[n + 1];
    for (int i0 = b; i0 < e; i0 += 32) {
        int ii = i0 + lane;
        int   s_l = (ii < e) ? cs[ii] : 0;
        float w_l = (ii < e) ? cw[ii] : 0.0f;
        int cnt = min(32, e - i0);
        for (int j = 0; j < cnt; ++j) {
            int   s = __shfl(s_l, j, 32);
            float w = __shfl(w_l, j, 32);
            acc += w * base[(size_t)s * 64];
        }
    }
    agg[(size_t)n * 128 + k] = acc * invd[n];
}

// ---------------------------------------------------------------- WMMA GEMM
// out[M,NCOLS] = concat_K(s0[:,0:w0], s1[:,0:w1], s2[:,rest]) (f32->bf16) @ Wb + bias
// Wb is in fragment-major layout (see pack_wmma_b). Block: 32 rows x NCOLS; 8 waves.
// M must be a multiple of 32 (holds: 20000, 240000).
template <int K, int NCOLS, bool RELU>
__global__ __launch_bounds__(256) void gemm_wmma_kernel(
    const float* __restrict__ s0, int w0, int ld0,
    const float* __restrict__ s1, int w1, int ld1,
    const float* __restrict__ s2, int ld2,
    const __bf16* __restrict__ Wb, const float* __restrict__ bias,
    float* __restrict__ outp, float* __restrict__ relu_outp) {
    __shared__ __bf16 sA[32][K + 8];
    const int tid = threadIdx.x;
    const int rowBase = blockIdx.x * 32;

    // Stage A tiles into LDS, converting f32 -> bf16
    for (int i = tid; i < 32 * K; i += 256) {
        int r = i / K, k = i - r * K;
        int grow = rowBase + r;
        float v;
        if (k < w0)            v = s0[(size_t)grow * ld0 + k];
        else if (k < w0 + w1)  v = s1[(size_t)grow * ld1 + (k - w0)];
        else                   v = s2[(size_t)grow * ld2 + (k - w0 - w1)];
        sA[r][k] = (__bf16)v;
    }
    __syncthreads();

    const int wave = tid >> 5, lane = tid & 31;
    const int r16 = lane & 15, hi = lane >> 4;
    constexpr int CT = NCOLS / 16;
    for (int tile = wave; tile < 2 * CT; tile += 8) {
        int rt = tile / CT, ct = tile - rt * CT;
        v8f acc = {0.f, 0.f, 0.f, 0.f, 0.f, 0.f, 0.f, 0.f};
        const int col = ct * 16 + r16;
#pragma unroll
        for (int kb = 0; kb < K; kb += 32) {
            // A fragment (contiguous K order, matching packed B): two aligned 16B LDS vector loads.
            // lanes 0-15 : row=lane,    elems 0..15 -> K=kb+0..15
            // lanes 16-31: row=lane-16, elems 0..15 -> K=kb+16..31
            const v8bf* ap = (const v8bf*)&sA[rt * 16 + r16][kb + hi * 16];
            v8bf alo = ap[0];
            v8bf ahi = ap[1];  // +8 elements
            v16bf a = __builtin_shufflevector(alo, ahi, 0, 1, 2, 3, 4, 5, 6, 7,
                                              8, 9, 10, 11, 12, 13, 14, 15);
            // B fragment: one contiguous 32B load from fragment-major weights.
            v16bf b = *(const v16bf*)(Wb + (((size_t)(kb >> 5) * CT + ct) * 32 + lane) * 16);
            acc = __builtin_amdgcn_wmma_f32_16x16x32_bf16(false, a, false, b, (short)0, acc, false, false);
        }
        // C/D layout: VGPR r -> M = hi*8 + r, N = col
        const float bv = bias[col];
#pragma unroll
        for (int r = 0; r < 8; ++r) {
            int m = rowBase + rt * 16 + hi * 8 + r;
            float v = acc[r] + bv;
            outp[(size_t)m * NCOLS + col] = v;
            if (RELU) relu_outp[(size_t)m * NCOLS + col] = fmaxf(v, 0.0f);
        }
    }
}

// ---------------------------------------------------------------- LSTM elementwise
__global__ __launch_bounds__(256) void lstm_ew(const float* __restrict__ gates, float* __restrict__ h_buf,
                                               float* __restrict__ c_buf, float* __restrict__ hcopy, int N) {
    int idx = blockIdx.x * 256 + threadIdx.x;
    if (idx >= N * 64) return;
    int n = idx >> 6, j = idx & 63;
    const float* g = gates + (size_t)n * 256;
    float ig = g[j], fg = g[64 + j], gg = g[128 + j], og = g[192 + j];
    float c = sigm(fg) * c_buf[idx] + sigm(ig) * tanhf(gg);
    float hv = sigm(og) * tanhf(c);
    c_buf[idx] = c;
    h_buf[idx] = hv;
    if (hcopy) hcopy[idx] = hv;
}

// ---------------------------------------------------------------- output projection (128 -> 8, last 4 steps)
__global__ __launch_bounds__(256) void out_proj(const float* __restrict__ xr, const float* __restrict__ h2l4,
                                                const float* __restrict__ W_out, const float* __restrict__ b_out,
                                                float* __restrict__ out, int N) {
    int idx = blockIdx.x * 256 + threadIdx.x;
    const int total = 4 * N * 8;
    if (idx >= total) return;
    int tp = idx / (N * 8);
    int rem = idx - tp * N * 8;
    int n = rem >> 3, o = rem & 7;
    const float* xrt = xr + (size_t)(8 + tp) * N * 64 + (size_t)n * 64;  // t = T-4+tp = 8+tp
    const float* h2t = h2l4 + (size_t)tp * N * 64 + (size_t)n * 64;
    float acc = b_out[o];
#pragma unroll 8
    for (int k = 0; k < 64; ++k) acc += xrt[k] * W_out[k * 8 + o];
#pragma unroll 8
    for (int k = 0; k < 64; ++k) acc += h2t[k] * W_out[(64 + k) * 8 + o];
    out[idx] = acc;
}

// ---------------------------------------------------------------- launch
extern "C" void kernel_launch(void* const* d_in, const int* in_sizes, int n_in,
                              void* d_out, int out_size, void* d_ws, size_t ws_size,
                              hipStream_t stream) {
    (void)in_sizes; (void)n_in; (void)out_size; (void)ws_size;

    constexpr int T = 12, N = 20000, E = 400000, H = 64, O = 8;

    const float* x       = (const float*)d_in[0];
    const int*   eidx    = (const int*)d_in[1];
    const float* ew      = (const float*)d_in[2];
    const float* sWr     = (const float*)d_in[3];
    const float* sWroot  = (const float*)d_in[4];
    const float* sb      = (const float*)d_in[5];
    const float* l1Wr    = (const float*)d_in[6];
    const float* l1Wroot = (const float*)d_in[7];
    const float* l1b     = (const float*)d_in[8];
    const float* l2Wr    = (const float*)d_in[9];
    const float* l2Wroot = (const float*)d_in[10];
    const float* l2b     = (const float*)d_in[11];
    const float* Wout    = (const float*)d_in[12];
    const float* bout    = (const float*)d_in[13];

    const int* srcp = eidx;      // edge_index[0]
    const int* dstp = eidx + E;  // edge_index[1]

    // workspace carve-out
    char* ws = (char*)d_ws;
    size_t off = 0;
    auto carve = [&](size_t bytes) -> char* {
        char* p = ws + off;
        off += (bytes + 255) & ~(size_t)255;
        return p;
    };
    float* xr    = (float*)carve((size_t)T * N * H * 4);          // relu(emb), [T,N,64]
    float* saggs = (float*)carve((size_t)T * N * 32 * 4);         // SAGE aggregate; reused as agg+gates
    float* agg   = saggs;                                         // [N,128] (alias, after SAGE done)
    float* gates = saggs + (size_t)N * 128;                       // [N,256] (alias)
    float* hc    = (float*)carve((size_t)2 * N * H * 4);
    float* h_buf = hc;
    float* c_buf = hc + (size_t)N * H;
    float* h2l4  = (float*)carve((size_t)4 * N * H * 4);          // h2 for last 4 steps
    int*   deg     = (int*)carve((size_t)N * 4);
    int*   rowst   = (int*)carve((size_t)(N + 1) * 4);
    int*   cursor  = (int*)carve((size_t)N * 4);
    float* invdeg  = (float*)carve((size_t)N * 4);
    int*   csrc    = (int*)carve((size_t)E * 4);
    float* cw      = (float*)carve((size_t)E * 4);
    __bf16* WbS = (__bf16*)carve((size_t)64 * 64 * 2);
    __bf16* Wb1 = (__bf16*)carve((size_t)256 * 256 * 2);
    __bf16* Wb2 = (__bf16*)carve((size_t)256 * 256 * 2);

    float* outp    = (float*)d_out;                  // [4,N,8]
    float* c2_out  = outp + (size_t)4 * N * O;       // [N,64]
    float* emb_out = c2_out + (size_t)N * H;         // [T,N,64]

    // ---- CSR build (once; edge set is static across all 24 recurrent steps)
    zero_i32<<<(N + 255) / 256, 256, 0, stream>>>(deg, N);
    zero_i32<<<(N + 255) / 256, 256, 0, stream>>>(cursor, N);
    deg_count<<<(E + 255) / 256, 256, 0, stream>>>(dstp, deg, E);
    scan_deg<<<1, 1024, 0, stream>>>(deg, rowst, invdeg, N);
    csr_fill<<<(E + 255) / 256, 256, 0, stream>>>(srcp, dstp, ew, rowst, cursor, csrc, cw, E);

    // ---- stacked bf16 weights [Wr ; Wroot], pre-swizzled into WMMA fragment layout
    pack_wmma_b<<<(64 * 64 + 255) / 256, 256, 0, stream>>>(sWr, sWroot, WbS, 32, 32, 64);
    pack_wmma_b<<<(256 * 256 + 255) / 256, 256, 0, stream>>>(l1Wr, l1Wroot, Wb1, 128, 128, 256);
    pack_wmma_b<<<(256 * 256 + 255) / 256, 256, 0, stream>>>(l2Wr, l2Wroot, Wb2, 128, 128, 256);

    // ---- SAGE: gather over all T at once, then one WMMA GEMM (K=64, rows=T*N)
    sage_gather<<<(T * N) / 8, 256, 0, stream>>>(rowst, csrc, cw, invdeg, x, saggs, N, T * N);
    gemm_wmma_kernel<64, 64, true><<<(T * N) / 32, 256, 0, stream>>>(
        saggs, 32, 32, x, 32, 32, x, 32, WbS, sb, emb_out, xr);

    // ---- graph LSTM layer 1 (h0=c0=0)
    zero_f32<<<(2 * N * H + 255) / 256, 256, 0, stream>>>(hc, 2 * N * H);
    for (int t = 0; t < T; ++t) {
        const float* xrt = xr + (size_t)t * N * H;
        lstm_gather<<<N / 2, 256, 0, stream>>>(rowst, csrc, cw, invdeg, xrt, h_buf, agg, N);
        gemm_wmma_kernel<256, 256, false><<<N / 32, 256, 0, stream>>>(
            agg, 128, 128, xrt, 64, 64, h_buf, 64, Wb1, l1b, gates, nullptr);
        lstm_ew<<<(N * H) / 256, 256, 0, stream>>>(gates, h_buf, c_buf, nullptr, N);
    }

    // ---- graph LSTM layer 2 (continues from h1[-1], c1 already in h_buf/c_buf)
    for (int t = 0; t < T; ++t) {
        const float* xrt = xr + (size_t)t * N * H;
        float* hcopy = (t >= T - 4) ? (h2l4 + (size_t)(t - (T - 4)) * N * H) : nullptr;
        lstm_gather<<<N / 2, 256, 0, stream>>>(rowst, csrc, cw, invdeg, xrt, h_buf, agg, N);
        gemm_wmma_kernel<256, 256, false><<<N / 32, 256, 0, stream>>>(
            agg, 128, 128, xrt, 64, 64, h_buf, 64, Wb2, l2b, gates, nullptr);
        lstm_ew<<<(N * H) / 256, 256, 0, stream>>>(gates, h_buf, c_buf, hcopy, N);
    }

    // ---- outputs
    out_proj<<<(4 * N * O + 255) / 256, 256, 0, stream>>>(xr, h2l4, Wout, bout, outp, N);
    hipMemcpyAsync(c2_out, c_buf, (size_t)N * H * 4, hipMemcpyDeviceToDevice, stream);
}
